// HeteroGAT_231928234677
// MI455X (gfx1250) — matrix operand order repaired and checked by reading, not compiled, not run on hardware
//
#include <hip/hip_runtime.h>
#include <hip/hip_bf16.h>
#include <math.h>

#define NUSR 100000
#define NITM 50000
#define NEDG 500000
#define FIN  128
#define HID  64
#define FOUT 64
#define NHEAD 4
#define SLOPE 0.2f

typedef __attribute__((ext_vector_type(16))) _Float16 v16h;
typedef __attribute__((ext_vector_type(8)))  _Float16 v8h;
typedef __attribute__((ext_vector_type(8)))  float    v8f;

// ---------------------------------------------------------------------------
// WMMA GEMM + bias (+optional ReLU):  C[M,N] = act(A[M,K] @ W[K,N] + b[N])
// 8 waves / block; each wave computes a 16x64 output strip (4 accumulators,
// 4 WMMAs per K-step with a single shared A fragment). The 64-column W slice
// for this block is staged once into LDS as f16, pre-swizzled into WMMA
// B-fragment order (lane-major, 24-half stride -> 16B-aligned ds_load_b128,
// bank-conflict friendly). Requires M%16==0, N%64==0, K%32==0.
// ---------------------------------------------------------------------------
template<int KTP, int ACT>
__global__ __launch_bounds__(256)
void gemm_bias_wmma(const float* __restrict__ A, const float* __restrict__ W,
                    const float* __restrict__ bias, float* __restrict__ C,
                    int M, int N) {
    constexpr int NKS = KTP / 32;                 // K-steps
    __shared__ _Float16 sB[NKS * 4 * 32 * 24];    // 4 ntiles, 32 lanes, 16+8 pad

    const int lane = threadIdx.x & 31;
    const int wave = threadIdx.x >> 5;
    const int cg   = blockIdx.y * 64;             // column-group base
    const int mtile = blockIdx.x * 8 + wave;

    // ---- cooperative stage of W[:, cg:cg+64] into LDS, fragment-swizzled ----
    for (int idx = threadIdx.x; idx < KTP * 64; idx += 256) {
        const int k = idx >> 6;
        const int n = idx & 63;
        const float w = W[(size_t)k * N + cg + n];
        const int ks = k >> 5, kr = k & 31;
        const int hi = (kr >> 3) & 1, e = (kr & 7) | ((kr >> 4) << 3);
        const int nt = n >> 4;
        const int l  = (hi << 4) | (n & 15);
        sB[((((ks << 2) + nt) << 5) + l) * 24 + e] = (_Float16)w;
    }
    __syncthreads();

    if (mtile * 16 >= M) return;                  // wave-uniform, EXEC stays full

    const int hi    = (lane >= 16) ? 1 : 0;
    const int nlo   = lane & 15;
    const int row_a = mtile * 16 + nlo;

    v8f acc[4];
    #pragma unroll
    for (int nt = 0; nt < 4; ++nt) {
        const float bv = bias[cg + nt * 16 + nlo];
        #pragma unroll
        for (int r = 0; r < 8; ++r) acc[nt][r] = bv;
    }

    #pragma unroll
    for (int ks = 0; ks < NKS; ++ks) {
        // A fragment: 16-bit A 16x32 layout (ISA 7.12.2)
        v16h af;
        #pragma unroll
        for (int v = 0; v < 8; ++v) {
            const int kb = (ks << 5) + ((v < 4) ? 0 : 16) + (hi ? 8 : 0) + ((v & 3) << 1);
            const float2 t = *(const float2*)(A + (size_t)row_a * KTP + kb);
            af[2 * v]     = (_Float16)t.x;
            af[2 * v + 1] = (_Float16)t.y;
        }
        #pragma unroll
        for (int nt = 0; nt < 4; ++nt) {
            const int base = ((((ks << 2) + nt) << 5) + lane) * 24;
            const v8h b0 = *(const v8h*)&sB[base];
            const v8h b1 = *(const v8h*)&sB[base + 8];
            v16h bf;
            #pragma unroll
            for (int e = 0; e < 8; ++e) { bf[e] = b0[e]; bf[e + 8] = b1[e]; }
            acc[nt] = __builtin_amdgcn_wmma_f32_16x16x32_f16(
                          false, af, false, bf, (short)0, acc[nt], false, false);
        }
    }

    #pragma unroll
    for (int nt = 0; nt < 4; ++nt) {
        #pragma unroll
        for (int r = 0; r < 8; ++r) {
            const int rr = mtile * 16 + r + (hi ? 8 : 0);
            float v = acc[nt][r];
            if (ACT == 1) v = v > 0.f ? v : 0.f;
            C[(size_t)rr * N + cg + nt * 16 + nlo] = v;
        }
    }
}

// ---------------------------------------------------------------------------
// Order-preserving float <-> uint encoding for atomicMax-based segment max
// ---------------------------------------------------------------------------
__device__ __forceinline__ unsigned f2ord(float f) {
    unsigned u = __float_as_uint(f);
    return (u & 0x80000000u) ? ~u : (u | 0x80000000u);
}
__device__ __forceinline__ float ord2f(unsigned u) {
    return (u & 0x80000000u) ? __uint_as_float(u ^ 0x80000000u)
                             : __uint_as_float(~u);
}

// per-(edge,head) attention logit + leaky-relu + running segment max
template<int H, int C>
__global__ void edge_logits(const float* __restrict__ tsrc, const float* __restrict__ tdst,
                            const int* __restrict__ esrc, const int* __restrict__ edst,
                            const float* __restrict__ att,
                            float* __restrict__ alpha, unsigned* __restrict__ amax, int E) {
    const int t = blockIdx.x * blockDim.x + threadIdx.x;
    if (t >= E * H) return;
    const int e = t / H, h = t % H;
    const int s = esrc[e], d = edst[e];
    const float4* xi = (const float4*)(tdst + (size_t)d * H * C + h * C);
    const float4* xj = (const float4*)(tsrc + (size_t)s * H * C + h * C);
    const float4* ai = (const float4*)(att + h * 2 * C);
    const float4* aj = ai + C / 4;
    float a = 0.f;
    #pragma unroll 4
    for (int c = 0; c < C / 4; ++c) {
        const float4 vi = xi[c], va = ai[c], vj = xj[c], vb = aj[c];
        a += vi.x * va.x + vi.y * va.y + vi.z * va.z + vi.w * va.w;
        a += vj.x * vb.x + vj.y * vb.y + vj.z * vb.z + vj.w * vb.w;
    }
    a = a > 0.f ? a : SLOPE * a;
    alpha[t] = a;
    atomicMax(&amax[(size_t)d * H + h], f2ord(a));
}

// exp(a - max) and segment sum
template<int H>
__global__ void edge_expsum(const int* __restrict__ edst,
                            const unsigned* __restrict__ amax,
                            float* __restrict__ alpha, float* __restrict__ ssum, int E) {
    const int t = blockIdx.x * blockDim.x + threadIdx.x;
    if (t >= E * H) return;
    const int e = t / H, h = t % H;
    const int d = edst[e];
    const unsigned enc = amax[(size_t)d * H + h];
    const float m = enc ? ord2f(enc) : 0.f;   // empty segment -> 0 (matches ref)
    const float ex = expf(alpha[t] - m);
    alpha[t] = ex;
    atomicAdd(&ssum[(size_t)d * H + h], ex);
}

// normalize and scatter-add weighted source features into out
template<int H, int C>
__global__ void edge_scatter(const float* __restrict__ tsrc,
                             const int* __restrict__ esrc, const int* __restrict__ edst,
                             const float* __restrict__ alpha, const float* __restrict__ ssum,
                             float* __restrict__ out, int E) {
    const int t = blockIdx.x * blockDim.x + threadIdx.x;
    if (t >= E * H) return;
    const int e = t / H, h = t % H;
    const int s = esrc[e], d = edst[e];
    const float a = alpha[t] / (ssum[(size_t)d * H + h] + 1e-16f);
    const float4* xj = (const float4*)(tsrc + (size_t)s * H * C + h * C);
    float* o = out + (size_t)d * H * C + h * C;
    #pragma unroll 4
    for (int c = 0; c < C / 4; ++c) {
        const float4 v = xj[c];
        atomicAdd(&o[4 * c + 0], v.x * a);
        atomicAdd(&o[4 * c + 1], v.y * a);
        atomicAdd(&o[4 * c + 2], v.z * a);
        atomicAdd(&o[4 * c + 3], v.w * a);
    }
}

// mean over heads + ELU
template<int H>
__global__ void mean_elu(const float* __restrict__ in, float* __restrict__ out, int n) {
    const int t = blockIdx.x * blockDim.x + threadIdx.x;
    if (t >= n * 64) return;
    const int node = t / 64, c = t % 64;
    float s = 0.f;
    #pragma unroll
    for (int h = 0; h < H; ++h) s += in[(size_t)node * H * 64 + h * 64 + c];
    s *= (1.0f / H);
    out[t] = s > 0.f ? s : expm1f(s);
}

// ---------------------------------------------------------------------------
extern "C" void kernel_launch(void* const* d_in, const int* in_sizes, int n_in,
                              void* d_out, int out_size, void* d_ws, size_t ws_size,
                              hipStream_t stream) {
    (void)in_sizes; (void)n_in; (void)out_size; (void)ws_size;
    const float* x_user = (const float*)d_in[0];
    const float* x_item = (const float*)d_in[1];
    const int* eui_src  = (const int*)d_in[2];
    const int* eui_dst  = (const int*)d_in[3];
    const int* eiu_src  = (const int*)d_in[4];
    const int* eiu_dst  = (const int*)d_in[5];
    const float* W1u = (const float*)d_in[6];  const float* b1u = (const float*)d_in[7];
    const float* W1i = (const float*)d_in[8];  const float* b1i = (const float*)d_in[9];
    const float* att1_ui = (const float*)d_in[10];
    const float* att1_iu = (const float*)d_in[11];
    const float* W2u = (const float*)d_in[12]; const float* b2u = (const float*)d_in[13];
    const float* W2i = (const float*)d_in[14]; const float* b2i = (const float*)d_in[15];
    const float* att2_ui = (const float*)d_in[16];
    const float* att2_iu = (const float*)d_in[17];
    const float* Wc1 = (const float*)d_in[18]; const float* bc1 = (const float*)d_in[19];
    const float* Wc2 = (const float*)d_in[20]; const float* bc2 = (const float*)d_in[21];

    float* out = (float*)d_out;
    float* out_u   = out;                                 // (NU,64)
    float* out_it  = out + (size_t)NUSR * 64;             // (NI,64)
    float* out_cat = out + (size_t)(NUSR + NITM) * 64;    // (NI,64)

    // workspace layout (floats)
    float* ws = (float*)d_ws;
    float*    T1U   = ws;                                  // NU*256
    float*    T1I   = ws + (size_t)NUSR * 256;             // NI*256
    float*    OI    = T1I + (size_t)NITM * 256;            // NI*256 (ui-pass output)
    float*    ALPHA = OI  + (size_t)NITM * 256;            // E*NHEAD
    unsigned* AMAX  = (unsigned*)(ALPHA + (size_t)NEDG * NHEAD); // max(NU,NI)*NHEAD
    float*    SUM   = (float*)(AMAX + (size_t)NUSR * NHEAD);     // max(NU,NI)*NHEAD
    // layer-2 reuse of freed regions
    float* U1   = T1I;                         // NU*64 (T1I free after iu scatter)
    float* I1   = T1I + (size_t)NUSR * 64;     // NI*64
    float* T2U  = T1U;                         // NU*64 (T1U free after mean_elu)
    float* T2I  = T1U + (size_t)NUSR * 64;     // NI*64
    float* OI2  = T2I + (size_t)NITM * 64;     // NI*64
    float* HBUF = OI2 + (size_t)NITM * 64;     // NI*64 classifier hidden

    const int BLK = 256;
    const int gE1 = (NEDG * NHEAD + BLK - 1) / BLK;
    const int gE2 = (NEDG + BLK - 1) / BLK;
    const int gMU = (NUSR / 16 + 7) / 8;       // 8 M-tiles per block
    const int gMI = (NITM / 16 + 7) / 8;

    // ---- Layer 1 transforms (WMMA) ----
    gemm_bias_wmma<FIN, 0><<<dim3(gMU, 4), 256, 0, stream>>>(x_user, W1u, b1u, T1U, NUSR, NHEAD * HID);
    gemm_bias_wmma<FIN, 0><<<dim3(gMI, 4), 256, 0, stream>>>(x_item, W1i, b1i, T1I, NITM, NHEAD * HID);

    // ---- Layer 1, user->item pass (dst = items) ----
    hipMemcpyAsync(OI, T1I, (size_t)NITM * 256 * sizeof(float), hipMemcpyDeviceToDevice, stream);
    hipMemsetAsync(AMAX, 0, (size_t)NITM * NHEAD * sizeof(unsigned), stream);
    hipMemsetAsync(SUM,  0, (size_t)NITM * NHEAD * sizeof(float), stream);
    edge_logits<NHEAD, HID><<<gE1, BLK, 0, stream>>>(T1U, T1I, eui_src, eui_dst, att1_ui, ALPHA, AMAX, NEDG);
    edge_expsum<NHEAD><<<gE1, BLK, 0, stream>>>(eui_dst, AMAX, ALPHA, SUM, NEDG);
    edge_scatter<NHEAD, HID><<<gE1, BLK, 0, stream>>>(T1U, eui_src, eui_dst, ALPHA, SUM, OI, NEDG);

    // ---- Layer 1, item->user pass (dst = users), scatter into T1U in place ----
    hipMemsetAsync(AMAX, 0, (size_t)NUSR * NHEAD * sizeof(unsigned), stream);
    hipMemsetAsync(SUM,  0, (size_t)NUSR * NHEAD * sizeof(float), stream);
    edge_logits<NHEAD, HID><<<gE1, BLK, 0, stream>>>(T1I, T1U, eiu_src, eiu_dst, att1_iu, ALPHA, AMAX, NEDG);
    edge_expsum<NHEAD><<<gE1, BLK, 0, stream>>>(eiu_dst, AMAX, ALPHA, SUM, NEDG);
    edge_scatter<NHEAD, HID><<<gE1, BLK, 0, stream>>>(T1I, eiu_src, eiu_dst, ALPHA, SUM, T1U, NEDG);

    // ---- head mean + ELU ----
    mean_elu<NHEAD><<<((NUSR * 64) + BLK - 1) / BLK, BLK, 0, stream>>>(T1U, U1, NUSR);
    mean_elu<NHEAD><<<((NITM * 64) + BLK - 1) / BLK, BLK, 0, stream>>>(OI,  I1, NITM);

    // ---- Layer 2 transforms (WMMA, H=1) ----
    gemm_bias_wmma<HID, 0><<<dim3(gMU, 1), 256, 0, stream>>>(U1, W2u, b2u, T2U, NUSR, FOUT);
    gemm_bias_wmma<HID, 0><<<dim3(gMI, 1), 256, 0, stream>>>(I1, W2i, b2i, T2I, NITM, FOUT);

    // ---- Layer 2, user->item pass ----
    hipMemcpyAsync(OI2, T2I, (size_t)NITM * 64 * sizeof(float), hipMemcpyDeviceToDevice, stream);
    hipMemsetAsync(AMAX, 0, (size_t)NITM * sizeof(unsigned), stream);
    hipMemsetAsync(SUM,  0, (size_t)NITM * sizeof(float), stream);
    edge_logits<1, FOUT><<<gE2, BLK, 0, stream>>>(T2U, T2I, eui_src, eui_dst, att2_ui, ALPHA, AMAX, NEDG);
    edge_expsum<1><<<gE2, BLK, 0, stream>>>(eui_dst, AMAX, ALPHA, SUM, NEDG);
    edge_scatter<1, FOUT><<<gE2, BLK, 0, stream>>>(T2U, eui_src, eui_dst, ALPHA, SUM, OI2, NEDG);

    // ---- Layer 2, item->user pass (scatter into T2U in place) ----
    hipMemsetAsync(AMAX, 0, (size_t)NUSR * sizeof(unsigned), stream);
    hipMemsetAsync(SUM,  0, (size_t)NUSR * sizeof(float), stream);
    edge_logits<1, FOUT><<<gE2, BLK, 0, stream>>>(T2I, T2U, eiu_src, eiu_dst, att2_iu, ALPHA, AMAX, NEDG);
    edge_expsum<1><<<gE2, BLK, 0, stream>>>(eiu_dst, AMAX, ALPHA, SUM, NEDG);
    edge_scatter<1, FOUT><<<gE2, BLK, 0, stream>>>(T2I, eiu_src, eiu_dst, ALPHA, SUM, T2U, NEDG);

    // ---- final node outputs ----
    hipMemcpyAsync(out_u,  T2U, (size_t)NUSR * 64 * sizeof(float), hipMemcpyDeviceToDevice, stream);
    hipMemcpyAsync(out_it, OI2, (size_t)NITM * 64 * sizeof(float), hipMemcpyDeviceToDevice, stream);

    // ---- classifier head (WMMA): relu(it @ Wc1 + bc1) @ Wc2 + bc2 ----
    gemm_bias_wmma<FOUT, 1><<<dim3(gMI, 1), 256, 0, stream>>>(OI2,  Wc1, bc1, HBUF, NITM, HID);
    gemm_bias_wmma<HID, 0><<<dim3(gMI, 1), 256, 0, stream>>>(HBUF, Wc2, bc2, out_cat, NITM, 64);
}
